// KnotAttentionInvariant_6141803233389
// MI455X (gfx1250) — compile-verified, matrix operand order checked by reading, not run on hardware
//
#include <hip/hip_runtime.h>
#include <hip/hip_bf16.h>
#include <cstdint>

// ---- CDNA5 WMMA fragment types (wave32) ----
typedef __attribute__((ext_vector_type(16))) _Float16 v16h;
typedef __attribute__((ext_vector_type(8)))  float    v8f;

#define HEADS  4
#define CDIM   256           // input feature dim
#define NTYPES 7             // Q, K0, K1, K2, V0, V1, V2
#define NTILES 28            // NTYPES * HEADS (each tile = 64 output columns)
#define WSW_ELEMS (NTILES * 4 * 8 * 32 * 16)   // pre-swizzled f16 weight fragments

// order-preserving float <-> uint mapping for atomic max
__device__ __forceinline__ unsigned f2ord(float f) {
  unsigned u = __float_as_uint(f);
  return (u & 0x80000000u) ? ~u : (u | 0x80000000u);
}
__device__ __forceinline__ float ord2f(unsigned u) {
  return __uint_as_float((u & 0x80000000u) ? (u & 0x7FFFFFFFu) : ~u);
}

// ------------------------------------------------------------------
// Kernel 1: convert f32 weights -> f16, pre-swizzled into the exact
// per-lane B-fragment layout of V_WMMA_F32_16X16X32_F16 (B is 32x16 KxN;
// lane = N + 16*(K-half), VGPR v packs K = (v/4)*16 + half*8 + (v%4)*2).
// Wsw[((tile*4 + nb)*8 + ks)*32 + lane][0..15]
// ------------------------------------------------------------------
__global__ void prep_weights(const float* __restrict__ wq,
                             const float* __restrict__ wk,
                             const float* __restrict__ wv,
                             _Float16* __restrict__ Wsw) {
  int e = blockIdx.x * blockDim.x + threadIdx.x;
  if (e >= WSW_ELEMS) return;
  int per  = e & 15;            // element within lane's 16 halves
  int v    = per >> 1;          // VGPR index 0..7
  int p    = per & 1;           // element of the K-pair
  int lane = (e >> 4) & 31;
  int ks   = (e >> 9) & 7;      // K-step (32 per step)
  int nb   = (e >> 12) & 3;     // 16-wide N subtile within the 64-wide tile
  int tile = e >> 14;           // 0..27
  int t = tile >> 2, h = tile & 3;
  int K = ks * 32 + ((v >> 2) << 4) + ((lane >> 4) << 3) + ((v & 3) << 1) + p;
  int c = nb * 16 + (lane & 15);         // column within the 64-wide head block
  float val;
  if (t == 0)      val = wq[(((size_t)h)        * CDIM + K) * 64 + c];
  else if (t <= 3) val = wk[(((size_t)h * 3 + (t - 1)) * CDIM + K) * 64 + c];
  else             val = wv[(((size_t)h * 3 + (t - 4)) * CDIM + K) * 64 + c];
  Wsw[e] = (_Float16)val;
}

// ------------------------------------------------------------------
// Kernel 2: GEMM  P[t][n][h*64+c] = x[n][:] @ W_t_h[:, c]   (f16 out, f32 acc)
// Block = 8 waves; wave owns a 16(M) x 64(N) tile of one (t,h) column block.
// 32 v_wmma_f32_16x16x32_f16 per wave. EXEC all-1s (tail rows clamped).
// ------------------------------------------------------------------
__global__ void __launch_bounds__(256) proj_gemm(const float* __restrict__ x,
                                                 const _Float16* __restrict__ Wsw,
                                                 _Float16* __restrict__ P, int n) {
  const int lane = threadIdx.x & 31;
  const int wave = threadIdx.x >> 5;
  const int tile = blockIdx.y;                    // 0..27  -> (t,h)
  const int t = tile >> 2, h = tile & 3;
  const int rowBase = blockIdx.x * 128 + wave * 16;
  const int khalf = (lane >> 4) << 3;             // +8 for upper half-wave

  int arow = rowBase + (lane & 15);
  if (arow >= n) arow = n - 1;                    // clamp: keep EXEC all-1 for WMMA
  const float* __restrict__ xrow = x + (size_t)arow * CDIM;

  v8f acc0 = {}, acc1 = {}, acc2 = {}, acc3 = {};
  const _Float16* __restrict__ wb = Wsw + (size_t)tile * (4 * 8 * 32 * 16);

#pragma unroll
  for (int ks = 0; ks < 8; ++ks) {
    v16h a;
#pragma unroll
    for (int v = 0; v < 8; ++v) {
      const int K = ks * 32 + ((v >> 2) << 4) + khalf + ((v & 3) << 1);
      const float2 xv = *(const float2*)(xrow + K);   // 8B-aligned pair
      a[2 * v]     = (_Float16)xv.x;
      a[2 * v + 1] = (_Float16)xv.y;
    }
    const v16h b0 = *(const v16h*)(wb + ((size_t)((0 * 8 + ks) * 32 + lane)) * 16);
    const v16h b1 = *(const v16h*)(wb + ((size_t)((1 * 8 + ks) * 32 + lane)) * 16);
    const v16h b2 = *(const v16h*)(wb + ((size_t)((2 * 8 + ks) * 32 + lane)) * 16);
    const v16h b3 = *(const v16h*)(wb + ((size_t)((3 * 8 + ks) * 32 + lane)) * 16);
    acc0 = __builtin_amdgcn_wmma_f32_16x16x32_f16(false, a, false, b0, (short)0, acc0, false, false);
    acc1 = __builtin_amdgcn_wmma_f32_16x16x32_f16(false, a, false, b1, (short)0, acc1, false, false);
    acc2 = __builtin_amdgcn_wmma_f32_16x16x32_f16(false, a, false, b2, (short)0, acc2, false, false);
    acc3 = __builtin_amdgcn_wmma_f32_16x16x32_f16(false, a, false, b3, (short)0, acc3, false, false);
  }

  // D layout: lane L holds N=L%16, VGPR v holds M = v + 8*(L/16)
  _Float16* __restrict__ Pt = P + (size_t)t * n * CDIM;
  const int cbase = h * 64 + (lane & 15);
#pragma unroll
  for (int v = 0; v < 8; ++v) {
    const int orow = rowBase + v + khalf;
    if (orow < n) {
      _Float16* dst = Pt + (size_t)orow * CDIM + cbase;
      dst[0]  = (_Float16)acc0[v];
      dst[16] = (_Float16)acc1[v];
      dst[32] = (_Float16)acc2[v];
      dst[48] = (_Float16)acc3[v];
    }
  }
}

// ------------------------------------------------------------------
// Kernel 3: per (n,h) compute 5 logits = Q.K/8, store, and global max
// per (h,r) bin via LDS atomic-max then 20 global atomics per block.
// ------------------------------------------------------------------
__global__ void __launch_bounds__(256) logits_max(const _Float16* __restrict__ P,
                                                  const long long* __restrict__ nbr,
                                                  float* __restrict__ logits,
                                                  unsigned* __restrict__ gmax, int n) {
  __shared__ unsigned smax[20];
  const int tid = threadIdx.x;
  if (tid < 20) smax[tid] = 0u;
  __syncthreads();

  const long long gid = (long long)blockIdx.x * blockDim.x + tid;
  if (gid < (long long)n * HEADS) {
    const int nid = (int)(gid >> 2);
    const int h   = (int)(gid & 3);
    const size_t npc = (size_t)n * CDIM;
    const _Float16* __restrict__ q  = P + 0 * npc + (size_t)nid * CDIM + h * 64;
    const _Float16* __restrict__ k0 = P + 1 * npc + (size_t)nid * CDIM + h * 64;
    const _Float16* __restrict__ Pk1 = P + 2 * npc;
    const _Float16* __restrict__ Pk2 = P + 3 * npc;

    long long m[4];
#pragma unroll
    for (int j = 0; j < 4; ++j) {
      long long mm = nbr[(size_t)nid * 4 + j];
      if (mm < 0) mm = 0; else if (mm >= n) mm = n - 1;
      m[j] = mm;
      const _Float16* pre = ((j < 2) ? Pk1 : Pk2) + (size_t)mm * CDIM + h * 64;
      __builtin_prefetch(pre, 0, 1);               // global_prefetch_b8
    }

#pragma unroll
    for (int r = 0; r < 5; ++r) {
      const _Float16* __restrict__ krow =
          (r == 0) ? k0 : (((r <= 2) ? Pk1 : Pk2) + (size_t)m[r - 1] * CDIM + h * 64);
      float dot = 0.f;
#pragma unroll
      for (int c = 0; c < 4; ++c) {
        const v16h qc = *(const v16h*)(q + c * 16);
        const v16h kc = *(const v16h*)(krow + c * 16);
#pragma unroll
        for (int i = 0; i < 16; ++i) dot += (float)qc[i] * (float)kc[i];
      }
      const float a = dot * 0.125f;                // 1/sqrt(64)
      logits[(size_t)nid * 20 + h * 5 + r] = a;
      atomicMax(&smax[h * 5 + r], f2ord(a));
    }
  }
  __syncthreads();
  if (tid < 20 && smax[tid] != 0u) atomicMax(&gmax[tid], smax[tid]);
}

// ------------------------------------------------------------------
// Kernel 4: e = exp(logit - gmax[bin]); store back; global sum per bin.
// ------------------------------------------------------------------
__global__ void __launch_bounds__(256) expsum(float* __restrict__ logits,
                                              const unsigned* __restrict__ gmax,
                                              float* __restrict__ gsum, int n) {
  __shared__ float ssum[20];
  const int tid = threadIdx.x;
  if (tid < 20) ssum[tid] = 0.f;
  __syncthreads();
  const long long gid = (long long)blockIdx.x * blockDim.x + tid;
  if (gid < (long long)n * 20) {
    const int bin = (int)(gid % 20);               // layout [n][h][r] -> bin = h*5+r
    const float e = __expf(logits[gid] - ord2f(gmax[bin]));
    logits[gid] = e;
    atomicAdd(&ssum[bin], e);
  }
  __syncthreads();
  if (tid < 20 && ssum[tid] != 0.f) atomicAdd(&gsum[tid], ssum[tid]);
}

// ------------------------------------------------------------------
// Kernel 5: Z[n,h,:] = sum_r (e_r / gsum[h,r]) * V_r ; write f32 output.
// ------------------------------------------------------------------
__global__ void __launch_bounds__(256) combine(const _Float16* __restrict__ P,
                                               const long long* __restrict__ nbr,
                                               const float* __restrict__ logits,
                                               const float* __restrict__ gsum,
                                               float* __restrict__ out, int n) {
  const long long gid = (long long)blockIdx.x * blockDim.x + threadIdx.x;
  if (gid >= (long long)n * HEADS) return;
  const int nid = (int)(gid >> 2);
  const int h   = (int)(gid & 3);
  const size_t npc = (size_t)n * CDIM;
  const _Float16* __restrict__ Pv1 = P + 5 * npc;
  const _Float16* __restrict__ Pv2 = P + 6 * npc;

  float w[5];
#pragma unroll
  for (int r = 0; r < 5; ++r)
    w[r] = logits[(size_t)nid * 20 + h * 5 + r] / gsum[h * 5 + r];

  const _Float16* vrow[5];
  vrow[0] = P + 4 * npc + (size_t)nid * CDIM + h * 64;
#pragma unroll
  for (int j = 0; j < 4; ++j) {
    long long mm = nbr[(size_t)nid * 4 + j];
    if (mm < 0) mm = 0; else if (mm >= n) mm = n - 1;
    vrow[1 + j] = ((j < 2) ? Pv1 : Pv2) + (size_t)mm * CDIM + h * 64;
    __builtin_prefetch(vrow[1 + j], 0, 1);
  }

  float* __restrict__ o = out + (size_t)nid * CDIM + h * 64;
#pragma unroll
  for (int c = 0; c < 4; ++c) {
    float z[16];
#pragma unroll
    for (int i = 0; i < 16; ++i) z[i] = 0.f;
#pragma unroll
    for (int r = 0; r < 5; ++r) {
      const v16h vc = *(const v16h*)(vrow[r] + c * 16);
#pragma unroll
      for (int i = 0; i < 16; ++i) z[i] += w[r] * (float)vc[i];
    }
#pragma unroll
    for (int i = 0; i < 16; ++i) o[c * 16 + i] = z[i];
  }
}

// ------------------------------------------------------------------
extern "C" void kernel_launch(void* const* d_in, const int* in_sizes, int n_in,
                              void* d_out, int out_size, void* d_ws, size_t ws_size,
                              hipStream_t stream) {
  const float*      x   = (const float*)d_in[0];
  const long long*  nbr = (const long long*)d_in[1];   // reference dtype int64
  const float*      wq  = (const float*)d_in[2];
  const float*      wk  = (const float*)d_in[3];
  const float*      wv  = (const float*)d_in[4];
  float*            out = (float*)d_out;
  const int n = in_sizes[0] / CDIM;

  // workspace carve-up (256B aligned sections)
  char* base = (char*)d_ws;
  size_t off = 0;
  auto take = [&](size_t bytes) -> char* {
    char* p = base + off;
    off = (off + bytes + 255) & ~(size_t)255;
    return p;
  };
  unsigned*  gmax   = (unsigned*)take(20 * sizeof(unsigned));
  float*     gsum   = (float*)take(20 * sizeof(float));
  _Float16*  Wsw    = (_Float16*)take((size_t)WSW_ELEMS * sizeof(_Float16));
  float*     logits = (float*)take((size_t)n * 20 * sizeof(float));
  _Float16*  Pproj  = (_Float16*)take((size_t)NTYPES * n * CDIM * sizeof(_Float16));
  (void)ws_size; (void)n_in; (void)out_size;

  // atomics accumulate -> must re-zero every launch (graph-replay safe)
  hipMemsetAsync(gmax, 0, 20 * sizeof(unsigned), stream);
  hipMemsetAsync(gsum, 0, 20 * sizeof(float), stream);

  prep_weights<<<(WSW_ELEMS + 255) / 256, 256, 0, stream>>>(wq, wk, wv, Wsw);

  dim3 gGemm((n + 127) / 128, NTILES);
  proj_gemm<<<gGemm, 256, 0, stream>>>(x, Wsw, Pproj, n);

  const long long nh = (long long)n * HEADS;
  logits_max<<<(unsigned)((nh + 255) / 256), 256, 0, stream>>>(Pproj, nbr, logits, gmax, n);

  const long long n20 = (long long)n * 20;
  expsum<<<(unsigned)((n20 + 255) / 256), 256, 0, stream>>>(logits, gmax, gsum, n);

  combine<<<(unsigned)((nh + 255) / 256), 256, 0, stream>>>(Pproj, nbr, logits, gsum, out, n);
}
